// SimpleHandwritingPredictionModel_38096359916087
// MI455X (gfx1250) — matrix operand order repaired and checked by reading, not compiled, not run on hardware
//
#include <hip/hip_runtime.h>
#include <hip/hip_bf16.h>

// ---------------- problem constants ----------------
#define UNITS   400
#define UP      416              // UNITS padded to multiple of 32 (WMMA K)
#define LDSP    424              // LDS row stride (bf16) to spread banks
#define G4      1600             // 4*UNITS
#define BATCH   64
#define TSTEPS  512
#define ROWS    (TSTEPS*BATCH)   // 32768 (row r = t*64 + b)
#define FEAT    3

typedef __bf16 bf16;
typedef __attribute__((ext_vector_type(8)))  __bf16 v8bf;
typedef __attribute__((ext_vector_type(16))) __bf16 v16bf;
typedef __attribute__((ext_vector_type(8)))  float  v8f;

__device__ __forceinline__ v16bf cat16(v8bf a, v8bf b) {
  return __builtin_shufflevector(a, b, 0,1,2,3,4,5,6,7,8,9,10,11,12,13,14,15);
}
__device__ __forceinline__ v8f wmma_bf16(v16bf a, v16bf b, v8f c) {
  return __builtin_amdgcn_wmma_f32_16x16x32_bf16(false, a, false, b, (short)0, c, false, false);
}
__device__ __forceinline__ float sigmoidf_(float x) {
  return 0.5f * tanhf(0.5f * x) + 0.5f;
}

// ---------------- utility kernels ----------------
__global__ void k_zero_u32(unsigned* p, long long n) {
  long long i = (long long)blockIdx.x * blockDim.x + threadIdx.x;
  if (i < n) p[i] = 0u;
}

// dst[n][k] = bf16(src[k][n]) for k<ksrc else 0 ; dst is [G4][UP]
__global__ void k_transpose_bf16(const float* __restrict__ src, bf16* __restrict__ dst, int ksrc) {
  int idx = blockIdx.x * blockDim.x + threadIdx.x;
  if (idx >= G4 * UP) return;
  int n = idx / UP, k = idx % UP;
  float v = (k < ksrc) ? src[(size_t)k * G4 + n] : 0.f;
  dst[idx] = (bf16)v;
}

// layer-0 input projection: x [B][T][3] -> xp [ROWS][G4], row r = t*64+b
__global__ void k_xproj0(const float* __restrict__ x, const float* __restrict__ Wx0,
                         const float* __restrict__ b0, float* __restrict__ xp) {
  long long idx = (long long)blockIdx.x * blockDim.x + threadIdx.x;
  if (idx >= (long long)ROWS * G4) return;
  int r = (int)(idx / G4), n = (int)(idx % G4);
  int t = r / BATCH, b = r % BATCH;
  const float* xr = x + ((size_t)b * TSTEPS + t) * FEAT;
  xp[idx] = b0[n] + xr[0]*Wx0[n] + xr[1]*Wx0[G4 + n] + xr[2]*Wx0[2*G4 + n];
}

// big GEMM: xp[r][n] = bias[n] + sum_k A[r][k]*BwT[n][k]
// A: [ROWS][UP] bf16, BwT: [G4][UP] bf16 (N-major transposed weight).
// One wave computes a 16x64 tile (1 m-tile x 4 n-tiles sharing the A fragment).
__global__ void __launch_bounds__(256)
k_xproj_gemm(const bf16* __restrict__ A, const bf16* __restrict__ BwT,
             const float* __restrict__ bias, float* __restrict__ xp) {
  int wid  = (blockIdx.x * blockDim.x + threadIdx.x) >> 5;   // 0..51199
  int lane = threadIdx.x & 31;
  int mt = wid / (G4/64);           // 0..2047
  int ng = wid % (G4/64);           // 0..24 -> 64 cols
  int nl  = lane & 15;
  int hi  = lane >> 4;
  int ako = hi * 16;
  int bko = hi * 16;
  const bf16* arow = A + ((size_t)(mt*16 + nl)) * UP + ako;
  v8f acc[4];
  for (int j = 0; j < 4; ++j) acc[j] = (v8f){0,0,0,0,0,0,0,0};
  for (int k0 = 0; k0 < UP; k0 += 32) {
    v8bf a0 = *(const v8bf*)(arow + k0);
    v8bf a1 = *(const v8bf*)(arow + k0 + 8);
    v16bf av = cat16(a0, a1);
#pragma unroll
    for (int j = 0; j < 4; ++j) {
      const bf16* brow = BwT + ((size_t)(ng*64 + j*16 + nl)) * UP + k0 + bko;
      v16bf bv = cat16(*(const v8bf*)brow, *(const v8bf*)(brow + 8));
      acc[j] = wmma_bf16(av, bv, acc[j]);
    }
  }
#pragma unroll
  for (int j = 0; j < 4; ++j) {
    int n = ng*64 + j*16 + nl;
    float bb = bias[n];
#pragma unroll
    for (int v = 0; v < 8; ++v) {
      int mm = mt*16 + v + 8*hi;
      xp[(size_t)mm * G4 + n] = acc[j][v] + bb;
    }
  }
}

// ---------------- persistent LSTM recurrence ----------------
// 25 blocks x 128 threads. Block = one 16-unit column tile (all 4 gates),
// wave = one 16-batch row tile. Wh slice cached in LDS across all 512 steps,
// cell state c lives in registers, one grid barrier per timestep.
__global__ void __launch_bounds__(128)
k_lstm(const float* __restrict__ xp, const bf16* __restrict__ WhT,
       const float* __restrict__ pi, const float* __restrict__ pf,
       const float* __restrict__ po, bf16* __restrict__ h_all,
       int* bar_count, int* bar_epoch) {
  __shared__ bf16 Bs[64][LDSP];
  int ut = blockIdx.x;                 // 0..24
  int u0 = ut * 16;
  int tid = threadIdx.x;
  int wave = tid >> 5;                 // m-tile 0..3
  int lane = tid & 31;

  // stage WhT rows (4 gates x 16 cols x UP K) into LDS, once
  for (int i = tid; i < 64 * (UP/8); i += blockDim.x) {
    int rr = i / (UP/8);
    int cc = (i % (UP/8)) * 8;
    int g = rr >> 4, j = rr & 15;
    const bf16* src = WhT + ((size_t)(g*UNITS + u0 + j)) * UP + cc;
    *(v8bf*)(&Bs[rr][cc]) = *(const v8bf*)src;
  }
  __syncthreads();

  int nl = lane & 15;
  int hi = lane >> 4;
  int ako = hi * 16;
  int bko = hi * 16;
  int ucol = u0 + nl;
  float vpi = pi[ucol], vpf = pf[ucol], vpo = po[ucol];
  float c[8];
#pragma unroll
  for (int v = 0; v < 8; ++v) c[v] = 0.f;

  int local_epoch = 0;
  for (int t = 0; t < TSTEPS; ++t) {
    // init gate accumulators from precomputed input projection
    v8f zi, zf, zg, zo;
    const float* xrow0 = xp + ((size_t)t * BATCH) * G4;
#pragma unroll
    for (int v = 0; v < 8; ++v) {
      int mm = wave*16 + v + 8*hi;
      const float* xr = xrow0 + (size_t)mm * G4;
      zi[v] = xr[0*UNITS + ucol];
      zf[v] = xr[1*UNITS + ucol];
      zg[v] = xr[2*UNITS + ucol];
      zo[v] = xr[3*UNITS + ucol];
    }
    if (t > 0) {
      const bf16* arow = h_all + ((size_t)(t-1)*BATCH + (wave*16 + nl)) * UP + ako;
      for (int k0 = 0; k0 < UP; k0 += 32) {
        v16bf av = cat16(*(const v8bf*)(arow + k0), *(const v8bf*)(arow + k0 + 8));
        { v16bf bv = cat16(*(const v8bf*)(&Bs[ 0 + nl][k0 + bko]), *(const v8bf*)(&Bs[ 0 + nl][k0 + bko + 8])); zi = wmma_bf16(av, bv, zi); }
        { v16bf bv = cat16(*(const v8bf*)(&Bs[16 + nl][k0 + bko]), *(const v8bf*)(&Bs[16 + nl][k0 + bko + 8])); zf = wmma_bf16(av, bv, zf); }
        { v16bf bv = cat16(*(const v8bf*)(&Bs[32 + nl][k0 + bko]), *(const v8bf*)(&Bs[32 + nl][k0 + bko + 8])); zg = wmma_bf16(av, bv, zg); }
        { v16bf bv = cat16(*(const v8bf*)(&Bs[48 + nl][k0 + bko]), *(const v8bf*)(&Bs[48 + nl][k0 + bko + 8])); zo = wmma_bf16(av, bv, zo); }
      }
    }
    // cell update entirely in registers (fragment layout: M = v + 8*hi, N = nl)
#pragma unroll
    for (int v = 0; v < 8; ++v) {
      float ig = sigmoidf_(zi[v] + vpi * c[v]);
      float fg = sigmoidf_(zf[v] + vpf * c[v]);
      float gg = tanhf(zg[v]);
      float cn = fg * c[v] + ig * gg;
      float og = sigmoidf_(zo[v] + vpo * cn);
      float hn = og * tanhf(cn);
      c[v] = cn;
      int mm = wave*16 + v + 8*hi;
      h_all[((size_t)t * BATCH + mm) * UP + ucol] = (bf16)hn;
    }
    // ---- software grid barrier (25 blocks, all resident) ----
    __threadfence();
    __syncthreads();
    if (tid == 0) {
      int e = local_epoch;
      int prev = atomicAdd(bar_count, 1);
      if (prev == (int)gridDim.x - 1) {
        *bar_count = 0;
        __threadfence();
        atomicAdd(bar_epoch, 1);
      }
      while (((volatile int*)bar_epoch)[0] <= e) { }
    }
    local_epoch++;
    __syncthreads();
    __threadfence();
  }
}

// final dense head: out[b][t][f] = bd[f] + sum_u h[t*64+b][u] * Wd[u][f]
__global__ void k_dense(const bf16* __restrict__ h_all, const float* __restrict__ Wd,
                        const float* __restrict__ bd, float* __restrict__ out) {
  int idx = blockIdx.x * blockDim.x + threadIdx.x;
  if (idx >= BATCH * TSTEPS * FEAT) return;
  int f = idx % FEAT;
  int bt = idx / FEAT;
  int b = bt / TSTEPS, t = bt % TSTEPS;
  const bf16* h = h_all + ((size_t)t * BATCH + b) * UP;
  float acc = bd[f];
  for (int u = 0; u < UNITS; ++u) acc += (float)h[u] * Wd[u * FEAT + f];
  out[idx] = acc;
}

// ---------------- host orchestration ----------------
extern "C" void kernel_launch(void* const* d_in, const int* in_sizes, int n_in,
                              void* d_out, int out_size, void* d_ws, size_t ws_size,
                              hipStream_t stream) {
  const float* x   = (const float*)d_in[0];
  const float* Wx[3] = { (const float*)d_in[1], (const float*)d_in[7],  (const float*)d_in[13] };
  const float* Wh[3] = { (const float*)d_in[2], (const float*)d_in[8],  (const float*)d_in[14] };
  const float* pi[3] = { (const float*)d_in[3], (const float*)d_in[9],  (const float*)d_in[15] };
  const float* pf[3] = { (const float*)d_in[4], (const float*)d_in[10], (const float*)d_in[16] };
  const float* po[3] = { (const float*)d_in[5], (const float*)d_in[11], (const float*)d_in[17] };
  const float* bb[3] = { (const float*)d_in[6], (const float*)d_in[12], (const float*)d_in[18] };
  const float* Wd = (const float*)d_in[19];
  const float* bd = (const float*)d_in[20];
  float* out = (float*)d_out;

  // workspace carve-up
  char* w = (char*)d_ws;
  size_t o = 0;
  float* xproj = (float*)(w + o); o += (size_t)ROWS * G4 * sizeof(float);        // 209.7 MB
  bf16*  h_all = (bf16*)(w + o);  o += (size_t)ROWS * UP * sizeof(bf16);         // 27.3 MB
  bf16*  WhT[3]; for (int l = 0; l < 3; ++l) { WhT[l] = (bf16*)(w + o); o += (size_t)G4 * UP * sizeof(bf16); }
  bf16*  WxT[3]; WxT[0] = nullptr;
  for (int l = 1; l < 3; ++l) { WxT[l] = (bf16*)(w + o); o += (size_t)G4 * UP * sizeof(bf16); }
  int* bars = (int*)(w + o); o += 64 * sizeof(int);   // per-layer {count, epoch}

  // zero h_all (K padding must be 0 for WMMA) and barrier words, every call
  {
    long long nw = ((long long)ROWS * UP * sizeof(bf16)) / 4;
    k_zero_u32<<<(unsigned)((nw + 255) / 256), 256, 0, stream>>>((unsigned*)h_all, nw);
    k_zero_u32<<<1, 64, 0, stream>>>((unsigned*)bars, 64);
  }
  // weight prep: transpose + bf16 + K-pad
  for (int l = 0; l < 3; ++l)
    k_transpose_bf16<<<(G4*UP + 255)/256, 256, 0, stream>>>(Wh[l], WhT[l], UNITS);
  for (int l = 1; l < 3; ++l)
    k_transpose_bf16<<<(G4*UP + 255)/256, 256, 0, stream>>>(Wx[l], WxT[l], UNITS);

  // layer 0 input projection (K=3, scalar)
  {
    long long n = (long long)ROWS * G4;
    k_xproj0<<<(unsigned)((n + 255) / 256), 256, 0, stream>>>(x, Wx[0], bb[0], xproj);
  }

  const int gemm_waves  = (ROWS/16) * (G4/64);       // 51200
  const int gemm_blocks = gemm_waves / 8;            // 6400 (256 thr = 8 waves)

  for (int l = 0; l < 3; ++l) {
    if (l > 0)  // xproj_l = h_{l-1} @ Wx_l + b_l  (big WMMA GEMM over all timesteps)
      k_xproj_gemm<<<gemm_blocks, 256, 0, stream>>>(h_all, WxT[l], bb[l], xproj);
    k_lstm<<<UNITS/16, 128, 0, stream>>>(xproj, WhT[l], pi[l], pf[l], po[l],
                                         h_all, bars + 2*l, bars + 2*l + 1);
  }

  k_dense<<<(BATCH*TSTEPS*FEAT + 255)/256, 256, 0, stream>>>(h_all, Wd, bd, out);
  (void)in_sizes; (void)n_in; (void)out_size; (void)ws_size;
}
